// EnhancedAttentionV2_57019985822189
// MI455X (gfx1250) — matrix operand order repaired and checked by reading, not compiled, not run on hardware
//
#include <hip/hip_runtime.h>

#define SEQ    4096
#define HIDDEN 1024
#define NHEADS 16
#define HDIM   64
#define WINH   128   // WIN/2

typedef __attribute__((ext_vector_type(16))) _Float16 v16h;
typedef __attribute__((ext_vector_type(8)))  _Float16 v8h;
typedef __attribute__((ext_vector_type(4)))  _Float16 v4h;
typedef __attribute__((ext_vector_type(8)))  float    v8f;

// Build a 16-half WMMA fragment register set from two contiguous 8-half runs.
static __device__ __forceinline__ v16h ld_frag(const _Float16* p0, const _Float16* p1) {
  v8h lo = *(const v8h*)p0;
  v8h hi = *(const v8h*)p1;
  return __builtin_shufflevector(lo, hi, 0,1,2,3,4,5,6,7,8,9,10,11,12,13,14,15);
}

// ---- CDNA5 async global->LDS copy (ASYNCcnt-tracked), ISA §10 / §15.18.3 ----
static __device__ __forceinline__ void async_copy_b128(uint32_t lds_addr, const void* gptr) {
  asm volatile("global_load_async_to_lds_b128 %0, %1, off"
               :: "v"(lds_addr), "v"(gptr)
               : "memory");
}
static __device__ __forceinline__ void wait_async0() {
  asm volatile("s_wait_asynccnt 0x0" ::: "memory");
}
static __device__ __forceinline__ uint32_t lds_addr_of(const void* p) {
  // generic (flat) pointers to LDS carry the byte offset in the low 32 bits
  return (uint32_t)(uintptr_t)p;
}

// ---------------------------------------------------------------- f32 -> f16
__global__ void __launch_bounds__(256) cvt_f32_to_f16(const float* __restrict__ src,
                                                      _Float16* __restrict__ dst) {
  size_t i = ((size_t)blockIdx.x * 256 + threadIdx.x) * 4;
  float4 v = *(const float4*)(src + i);
  v4h o;
  o[0] = (_Float16)v.x; o[1] = (_Float16)v.y; o[2] = (_Float16)v.z; o[3] = (_Float16)v.w;
  *(v4h*)(dst + i) = o;
}

// -------------------------------------- f32 -> f16 with transpose (weights)
__global__ void __launch_bounds__(256) cvt_transpose_f16(const float* __restrict__ W,
                                                         _Float16* __restrict__ Wt) {
  __shared__ _Float16 tile[32][33];
  const int k0 = blockIdx.x * 32;
  const int n0 = blockIdx.y * 32;
  const int t  = threadIdx.x;
  {
    const int kr = t >> 3;          // 0..31
    const int nc = (t & 7) << 2;    // 0..28
    float4 v = *(const float4*)(W + (size_t)(k0 + kr) * HIDDEN + n0 + nc);
    tile[nc + 0][kr] = (_Float16)v.x;
    tile[nc + 1][kr] = (_Float16)v.y;
    tile[nc + 2][kr] = (_Float16)v.z;
    tile[nc + 3][kr] = (_Float16)v.w;
  }
  __syncthreads();
  {
    const int nr = t >> 3;
    const int kc = (t & 7) << 2;
    v4h o;
    o[0] = tile[nr][kc + 0]; o[1] = tile[nr][kc + 1];
    o[2] = tile[nr][kc + 2]; o[3] = tile[nr][kc + 3];
    *(v4h*)(Wt + (size_t)(n0 + nr) * HIDDEN + k0 + kc) = o;
  }
}

// ------------------------------------------------- QKV projection GEMM (WMMA)
// C[4096,1024] = A x W + bias ; f16 in, f32 acc, f16 out.
// Block tile 128x128, 8 waves as 4(M)x2(N), wave tile 32x64 (2x4 WMMA frags).
// A tile double-buffered via async global->LDS; B frags direct from transposed
// Wt. k-loop manually unrolled by 2 + last two tiles peeled -> branch-free
// bodies so accumulators stay pinned in-place across WMMAs (no PHI copies).
__global__ void __launch_bounds__(256) qkv_gemm_f16_wmma(
    const _Float16* __restrict__ A, const _Float16* __restrict__ Wt,
    const float* __restrict__ bias, _Float16* __restrict__ out) {
  __shared__ __align__(16) _Float16 ldsA[2][128][32];   // [buf][m][k], 2 x 8KB
  const int tid    = threadIdx.x;
  const int lane   = tid & 31;
  const int wave   = tid >> 5;
  const int lane16 = lane & 15;
  const int laneHi = lane >> 4;
  const int m0 = blockIdx.x * 128;
  const int n0 = blockIdx.y * 128;
  const int wm = (wave >> 1) * 32;
  const int wn = (wave & 1) * 64;

  v8f acc[2][4] = {};

  const int ar = tid >> 1;          // A-tile row this thread stages
  const int ac = (tid & 1) << 4;    // A-tile col base (16 halves)
  const _Float16* aSrcBase = A + (size_t)(m0 + ar) * HIDDEN + ac;
  const uint32_t l0 = lds_addr_of(&ldsA[0][ar][ac]);
  const uint32_t l1 = lds_addr_of(&ldsA[1][ar][ac]);

  const _Float16* bBase = Wt + (size_t)(n0 + wn + lane16) * HIDDEN + laneHi * 8;

  auto stage = [&](int kt, uint32_t l) {
    const _Float16* src = aSrcBase + kt * 32;
    async_copy_b128(l,      src);
    async_copy_b128(l + 16, src + 8);
  };

  auto compute = [&](int buf, int k0) {
    #pragma unroll
    for (int mi = 0; mi < 2; ++mi) {
      const _Float16* ap = &ldsA[buf][wm + mi * 16 + lane16][laneHi * 8];
      v16h af = ld_frag(ap, ap + 16);
      #pragma unroll
      for (int ni = 0; ni < 4; ++ni) {
        const _Float16* bp = bBase + (size_t)ni * 16 * HIDDEN + k0;
        v16h bf = ld_frag(bp, bp + 16);
        acc[mi][ni] = __builtin_amdgcn_wmma_f32_16x16x32_f16(
            false, af, false, bf, (short)0, acc[mi][ni], false, false);
      }
    }
  };

  const int NT = HIDDEN / 32;       // 32 k-tiles
  stage(0, l0);
  for (int kt = 0; kt < NT - 2; kt += 2) {
    wait_async0();
    __syncthreads();
    stage(kt + 1, l1);
    compute(0, kt * 32);
    wait_async0();
    __syncthreads();
    stage(kt + 2, l0);
    compute(1, (kt + 1) * 32);
  }
  wait_async0();
  __syncthreads();
  stage(NT - 1, l1);
  compute(0, (NT - 2) * 32);
  wait_async0();
  __syncthreads();
  compute(1, (NT - 1) * 32);

  #pragma unroll
  for (int mi = 0; mi < 2; ++mi)
    #pragma unroll
    for (int ni = 0; ni < 4; ++ni) {
      const int n = n0 + wn + ni * 16 + lane16;
      const float bv = bias[n];
      #pragma unroll
      for (int r = 0; r < 8; ++r) {
        const int m = m0 + wm + mi * 16 + r + 8 * laneHi;
        out[(size_t)m * HIDDEN + n] = (_Float16)(acc[mi][ni][r] + bv);
      }
    }
}

// --------------------------------------- sliding-window flash attention (WMMA)
// grid = (NHEADS, SEQ/128); 8 waves, each owns 16 queries. All waves sweep the
// shared 384-key union window [q0-128, q0+256) in 12 tiles of 32 keys.
__global__ void __launch_bounds__(256) swa_attn_wmma(
    const _Float16* __restrict__ Q, const _Float16* __restrict__ K,
    const _Float16* __restrict__ V, float* __restrict__ ctx) {
  __shared__ __align__(16) _Float16 ldsVt[HDIM][32];    // V^T tile: [d][j]
  __shared__ __align__(16) _Float16 ldsP[8][16][32];    // per-wave P: [m][j]
  const int tid    = threadIdx.x;
  const int lane   = tid & 31;
  const int wave   = tid >> 5;
  const int lane16 = lane & 15;
  const int laneHi = lane >> 4;
  const int head = blockIdx.x;
  const int q0   = blockIdx.y * 128;
  const int qw   = q0 + wave * 16;
  const size_t hoff = (size_t)head * HDIM;

  // Q A-fragments: two K=32 chunks covering HDIM=64
  v16h qf[2];
  {
    const _Float16* qrow = Q + (size_t)(qw + lane16) * HIDDEN + hoff;
    #pragma unroll
    for (int c = 0; c < 2; ++c) {
      const _Float16* p = qrow + c * 32 + laneHi * 8;
      qf[c] = ld_frag(p, p + 16);
    }
  }

  float Mr[8], Lr[8];
  #pragma unroll
  for (int r = 0; r < 8; ++r) { Mr[r] = -1e30f; Lr[r] = 0.0f; }
  v8f cacc[4] = {};

  const int vj = tid >> 3;          // 0..31 : key row staged by this thread
  const int vd = (tid & 7) << 3;    // 0..56 : d base (8 halves)

  for (int t = 0; t < 12; ++t) {
    const int j0 = q0 - 128 + t * 32;

    // ---- scores S = (Q K^T)/8 : two 16x16 tiles over this 32-key tile ----
    v8f s[2] = {};
    #pragma unroll
    for (int nc = 0; nc < 2; ++nc) {
      const int jg = j0 + nc * 16 + lane16;
      const int jc = jg < 0 ? 0 : (jg >= SEQ ? SEQ - 1 : jg);
      const _Float16* krow = K + (size_t)jc * HIDDEN + hoff;
      // unconditional (clamped, non-faulting) warm of next K tile
      const int jn = j0 + 32 + nc * 16 + lane16;
      const int jnc = jn < 0 ? 0 : (jn >= SEQ ? SEQ - 1 : jn);
      __builtin_prefetch(K + (size_t)jnc * HIDDEN + hoff, 0, 0);
      #pragma unroll
      for (int c = 0; c < 2; ++c) {
        const _Float16* p = krow + c * 32 + laneHi * 8;    // table-order B frag runs
        v16h bf = ld_frag(p, p + 16);
        s[nc] = __builtin_amdgcn_wmma_f32_16x16x32_f16(
            false, qf[c], false, bf, (short)0, s[nc], false, false);
      }
    }

    // ---- scale, window mask (-10000, as in reference), online softmax ----
    float pt[2][8];
    #pragma unroll
    for (int r = 0; r < 8; ++r) {
      const int ig = qw + r + 8 * laneHi;
      float mx = -1e30f;
      #pragma unroll
      for (int nc = 0; nc < 2; ++nc) {
        const int jg = j0 + nc * 16 + lane16;
        float sv = s[nc][r] * 0.125f;
        const bool keep = (jg >= 0) && (jg < SEQ) && (jg >= ig - WINH) && (jg < ig + WINH);
        sv = keep ? sv : -10000.0f;
        pt[nc][r] = sv;
        mx = fmaxf(mx, sv);
      }
      #pragma unroll
      for (int off = 1; off < 16; off <<= 1)
        mx = fmaxf(mx, __shfl_xor(mx, off, 16));          // row lives in 16 lanes
      const float Mn = fmaxf(Mr[r], mx);
      const float alpha = __expf(Mr[r] - Mn);
      Mr[r] = Mn;
      float rs = 0.0f;
      #pragma unroll
      for (int nc = 0; nc < 2; ++nc) {
        const float p = __expf(pt[nc][r] - Mn);
        pt[nc][r] = p;
        rs += p;
      }
      #pragma unroll
      for (int off = 1; off < 16; off <<= 1)
        rs += __shfl_xor(rs, off, 16);
      Lr[r] = Lr[r] * alpha + rs;
      #pragma unroll
      for (int dc = 0; dc < 4; ++dc) cacc[dc][r] *= alpha;
    }

    // ---- stage V^T (shared) and P (per-wave) in LDS ----
    __syncthreads();   // previous-iteration LDS reads complete
    {
      const int jg = j0 + vj;
      const int jc = jg < 0 ? 0 : (jg >= SEQ ? SEQ - 1 : jg);
      const _Float16* vrow = V + (size_t)jc * HIDDEN + hoff + vd;
      v8h vv = *(const v8h*)vrow;
      #pragma unroll
      for (int i = 0; i < 8; ++i) ldsVt[vd + i][vj] = vv[i];
    }
    #pragma unroll
    for (int nc = 0; nc < 2; ++nc)
      #pragma unroll
      for (int r = 0; r < 8; ++r)
        ldsP[wave][r + 8 * laneHi][nc * 16 + lane16] = (_Float16)pt[nc][r];
    __syncthreads();

    // ---- context += P x V : 4 WMMAs across d-chunks ----
    {
      const _Float16* pp = &ldsP[wave][lane16][laneHi * 8];
      v16h pf = ld_frag(pp, pp + 16);                      // A frag (K=32 keys)
      #pragma unroll
      for (int dc = 0; dc < 4; ++dc) {
        const _Float16* vp = &ldsVt[dc * 16 + lane16][laneHi * 8];
        v16h bf = ld_frag(vp, vp + 16);                    // B frag
        cacc[dc] = __builtin_amdgcn_wmma_f32_16x16x32_f16(
            false, pf, false, bf, (short)0, cacc[dc], false, false);
      }
    }
  }

  // ---- normalize and store context (f32) ----
  #pragma unroll
  for (int dc = 0; dc < 4; ++dc)
    #pragma unroll
    for (int r = 0; r < 8; ++r) {
      const int m = qw + r + 8 * laneHi;
      const int d = dc * 16 + lane16;
      ctx[(size_t)m * HIDDEN + hoff + d] = cacc[dc][r] / Lr[r];
    }
}

// ---------------------------------------------------- residual + LayerNorm
__global__ void __launch_bounds__(256) residual_layernorm(
    const float* __restrict__ hs, const float* __restrict__ ctx,
    const float* __restrict__ gamma, const float* __restrict__ beta,
    float* __restrict__ out) {
  __shared__ float red[256];
  const int row = blockIdx.x, t = threadIdx.x;
  const size_t base = (size_t)row * HIDDEN + t * 4;
  float4 a = *(const float4*)(hs + base);
  float4 b = *(const float4*)(ctx + base);
  const float x0 = a.x + b.x, x1 = a.y + b.y, x2 = a.z + b.z, x3 = a.w + b.w;
  red[t] = x0 + x1 + x2 + x3;
  __syncthreads();
  for (int off = 128; off > 0; off >>= 1) {
    if (t < off) red[t] += red[t + off];
    __syncthreads();
  }
  const float mu = red[0] * (1.0f / HIDDEN);
  __syncthreads();
  const float d0 = x0 - mu, d1 = x1 - mu, d2 = x2 - mu, d3 = x3 - mu;
  red[t] = d0 * d0 + d1 * d1 + d2 * d2 + d3 * d3;
  __syncthreads();
  for (int off = 128; off > 0; off >>= 1) {
    if (t < off) red[t] += red[t + off];
    __syncthreads();
  }
  const float inv = rsqrtf(red[0] * (1.0f / HIDDEN) + 1e-12f);
  const float4 g  = *(const float4*)(gamma + t * 4);
  const float4 be = *(const float4*)(beta + t * 4);
  float4 o;
  o.x = d0 * inv * g.x + be.x;
  o.y = d1 * inv * g.y + be.y;
  o.z = d2 * inv * g.z + be.z;
  o.w = d3 * inv * g.w + be.w;
  *(float4*)(out + base) = o;
}

// ----------------------------------------------------------------- launcher
extern "C" void kernel_launch(void* const* d_in, const int* in_sizes, int n_in,
                              void* d_out, int out_size, void* d_ws, size_t ws_size,
                              hipStream_t stream) {
  const float* hs    = (const float*)d_in[0];
  const float* Wq    = (const float*)d_in[1];
  const float* bq    = (const float*)d_in[2];
  const float* Wk    = (const float*)d_in[3];
  const float* bk    = (const float*)d_in[4];
  const float* Wv    = (const float*)d_in[5];
  const float* bv    = (const float*)d_in[6];
  const float* gamma = (const float*)d_in[7];
  const float* beta  = (const float*)d_in[8];
  float* out = (float*)d_out;

  // workspace partition (~54 MB total)
  char* w = (char*)d_ws;
  _Float16* hs16  = (_Float16*)w; w += (size_t)SEQ * HIDDEN * sizeof(_Float16);
  _Float16* Wqt16 = (_Float16*)w; w += (size_t)HIDDEN * HIDDEN * sizeof(_Float16);
  _Float16* Wkt16 = (_Float16*)w; w += (size_t)HIDDEN * HIDDEN * sizeof(_Float16);
  _Float16* Wvt16 = (_Float16*)w; w += (size_t)HIDDEN * HIDDEN * sizeof(_Float16);
  _Float16* q16   = (_Float16*)w; w += (size_t)SEQ * HIDDEN * sizeof(_Float16);
  _Float16* k16   = (_Float16*)w; w += (size_t)SEQ * HIDDEN * sizeof(_Float16);
  _Float16* v16   = (_Float16*)w; w += (size_t)SEQ * HIDDEN * sizeof(_Float16);
  float*    ctx   = (float*)w;    w += (size_t)SEQ * HIDDEN * sizeof(float);

  // 1) conversions: hidden plain, weights transposed for direct B-frag loads
  cvt_f32_to_f16<<<(SEQ * HIDDEN) / (256 * 4), 256, 0, stream>>>(hs, hs16);
  dim3 tgrid(HIDDEN / 32, HIDDEN / 32);
  cvt_transpose_f16<<<tgrid, 256, 0, stream>>>(Wq, Wqt16);
  cvt_transpose_f16<<<tgrid, 256, 0, stream>>>(Wk, Wkt16);
  cvt_transpose_f16<<<tgrid, 256, 0, stream>>>(Wv, Wvt16);

  // 2) Q/K/V projections
  dim3 ggrid(SEQ / 128, HIDDEN / 128);
  qkv_gemm_f16_wmma<<<ggrid, 256, 0, stream>>>(hs16, Wqt16, bq, q16);
  qkv_gemm_f16_wmma<<<ggrid, 256, 0, stream>>>(hs16, Wkt16, bk, k16);
  qkv_gemm_f16_wmma<<<ggrid, 256, 0, stream>>>(hs16, Wvt16, bv, v16);

  // 3) sliding-window attention
  dim3 agrid(NHEADS, SEQ / 128);
  swa_attn_wmma<<<agrid, 256, 0, stream>>>(q16, k16, v16, ctx);

  // 4) residual + LayerNorm
  residual_layernorm<<<SEQ, 256, 0, stream>>>(hs, ctx, gamma, beta, out);
}
